// DialogueGNNModel_60112362275657
// MI455X (gfx1250) — matrix operand order repaired and checked by reading, not compiled, not run on hardware
//
#include <hip/hip_runtime.h>
#include <hip/hip_bf16.h>
#include <stdint.h>

#define L_ 256
#define B_ 64
#define D_ 1024
#define H_ 512
#define R_ 8
#define C_ 6
#define WIN_ 10
#define N_ (B_*L_)      // 16384
#define DH_ (D_+H_)     // 1536

typedef __attribute__((ext_vector_type(16))) __bf16 v16bf;
typedef __attribute__((ext_vector_type(8)))  float  v8f;

__device__ __forceinline__ unsigned short f2bf(float f){
  union { float f; unsigned u; } x; x.f = f;
  unsigned r = x.u + 0x7fffu + ((x.u >> 16) & 1u);
  return (unsigned short)(r >> 16);
}
__device__ __forceinline__ float bf2f(unsigned short h){
  union { unsigned u; float f; } x; x.u = ((unsigned)h) << 16;
  return x.f;
}

// ---------------- elementwise helpers ----------------

__global__ void k_f2bf(const float* __restrict__ s, unsigned short* __restrict__ d, int n){
  int i = blockIdx.x * 256 + threadIdx.x;
  if (i < n) d[i] = f2bf(s[i]);
}

// features [L,B,D] (s-major) -> x_bf [N,D] conversation-major (n = b*L+s), bf16
__global__ void k_xpose(const float* __restrict__ f, unsigned short* __restrict__ x){
  size_t gid = (size_t)blockIdx.x * 256 + threadIdx.x;   // over N*D
  int d = (int)(gid & (D_ - 1));
  int n = (int)(gid >> 10);
  int b = n >> 8, s = n & 255;
  x[gid] = f2bf(f[((size_t)(s * B_ + b) << 10) + d]);
}

// em_bf[n, 0:1024] = x_bf[n], em_bf[n, 1024:1536] = bf16(h2[n])
__global__ void k_empack(const unsigned short* __restrict__ xbf, const float* __restrict__ h2,
                         unsigned short* __restrict__ em){
  size_t gid = (size_t)blockIdx.x * 256 + threadIdx.x;   // over N*DH
  int d = (int)(gid % DH_);
  int n = (int)(gid / DH_);
  em[gid] = (d < D_) ? xbf[((size_t)n << 10) + d]
                     : f2bf(h2[((size_t)n << 9) + (d - D_)]);
}

// ---------------- WMMA GEMM (bf16 in, f32 acc) ----------------
// C = act( [C +] A*B + bias ), A:[M,K] row-major bf16, B:[K,N] (or [N,K] if TRANSB) bf16.
// 128x128 block tile, BK=32, 8 waves (4Mx2N), each wave 32x64 (2x4 fragments, 8 WMMA),
// double-buffered LDS with global prefetch overlapped with WMMA compute.

#define FB_TRANSB 1
#define FB_ACC    2
#define FB_RELU   4
#define FB_OUTBF  8
#define FB_BIAS  16

#define BMT 128
#define BNT 128
#define BKT 32
#define LROW 40   // shorts per LDS row: 32 data + 8 pad (keeps 16B alignment)

__global__ __launch_bounds__(256) void k_gemm(
    const unsigned short* __restrict__ A,
    const unsigned short* __restrict__ Bp,
    void* __restrict__ Cp, const float* __restrict__ bias,
    int M, int N, int K, int lda, int ldb, int ldc,
    long sA, long sB, long sC, int flags)
{
  __shared__ __align__(16) unsigned short As[2 * BMT * LROW];
  __shared__ __align__(16) unsigned short Bs[2 * BNT * LROW];
  const int bz = blockIdx.z;
  const unsigned short* Ab = A  + (size_t)bz * sA;
  const unsigned short* Bb = Bp + (size_t)bz * sB;
  const int tid  = threadIdx.x;
  const int lane = tid & 31;
  const int wave = tid >> 5;
  const int waveM = wave >> 1;            // 0..3 -> M groups of 32
  const int waveN = wave & 1;             // 0..1 -> N groups of 64
  const int tileM = blockIdx.y * BMT;
  const int tileN = blockIdx.x * BNT;
  const int l16 = lane & 15;
  const int lhi = (lane >> 4) & 1;

  v8f acc[2][4];
  const v8f vzero = {0.f,0.f,0.f,0.f,0.f,0.f,0.f,0.f};
  #pragma unroll
  for (int fm = 0; fm < 2; fm++)
    #pragma unroll
    for (int fn = 0; fn < 4; fn++) acc[fm][fn] = vzero;

  // global->register staging maps
  const int ar  = tid >> 1;            // 0..127 (row for A / n-row for transB B)
  const int ac  = (tid & 1) << 4;      // 0 or 16 (k column)
  const int bkr = tid >> 3;            // 0..31   (k row, non-trans B)
  const int bn0 = (tid & 7) << 4;      // 0..112  (n column, non-trans B)

  uint4 ra0, ra1, rb0, rb1;

  auto g_load = [&](int kb) {
    const uint4 z = make_uint4(0,0,0,0);
    ra0 = z; ra1 = z; rb0 = z; rb1 = z;
    int gr = tileM + ar;
    if (gr < M) {
      const uint4* s = (const uint4*)(Ab + (size_t)gr * lda + kb + ac);
      ra0 = s[0]; ra1 = s[1];
    }
    if (flags & FB_TRANSB) {
      int gn = tileN + ar;
      if (gn < N) {
        const uint4* s = (const uint4*)(Bb + (size_t)gn * ldb + kb + ac);
        rb0 = s[0]; rb1 = s[1];
      }
    } else {
      int gn = tileN + bn0;
      if (gn < N) {
        const uint4* s = (const uint4*)(Bb + (size_t)(kb + bkr) * ldb + gn);
        rb0 = s[0]; rb1 = s[1];
      }
    }
  };

  auto l_store = [&](int buf) {
    unsigned short* Ad = As + buf * (BMT * LROW) + ar * LROW + ac;
    ((uint4*)Ad)[0] = ra0; ((uint4*)Ad)[1] = ra1;
    if (flags & FB_TRANSB) {
      unsigned short* Bd = Bs + buf * (BNT * LROW) + ar * LROW + ac;
      ((uint4*)Bd)[0] = rb0; ((uint4*)Bd)[1] = rb1;
    } else {
      union { uint4 q[2]; unsigned short s[16]; } tb;
      tb.q[0] = rb0; tb.q[1] = rb1;
      unsigned short* Bbase = Bs + buf * (BNT * LROW);
      #pragma unroll
      for (int i = 0; i < 16; i++) Bbase[(bn0 + i) * LROW + bkr] = tb.s[i];
    }
  };

  const int nk = K / BKT;
  g_load(0);
  l_store(0);

  for (int t = 0; t < nk; ++t) {
    __syncthreads();
    if (t + 1 < nk) g_load((t + 1) * BKT);   // overlap next-tile fetch with compute

    const int buf = t & 1;
    union Frag { unsigned u[8]; v16bf v; };
    Frag fa[2], fb[4];
    #pragma unroll
    for (int fm = 0; fm < 2; fm++) {
      const unsigned short* base = As + buf * (BMT * LROW) + (waveM * 32 + fm * 16 + l16) * LROW;
      #pragma unroll
      for (int v = 0; v < 8; v++) {
        int kdw = ((v >> 2) << 3) + (lhi << 2) + (v & 3);   // ISA 16-bit A 16x32 layout
        fa[fm].u[v] = *(const unsigned*)(base + (kdw << 1));
      }
    }
    #pragma unroll
    for (int fn = 0; fn < 4; fn++) {
      const unsigned short* base = Bs + buf * (BNT * LROW) + (waveN * 64 + fn * 16 + l16) * LROW;
      #pragma unroll
      for (int v = 0; v < 8; v++) {
        int kdw = ((v >> 2) << 3) + (lhi << 2) + (v & 3);   // B 32x16: same K order as A (ISA 16-bit layout)
        fb[fn].u[v] = *(const unsigned*)(base + (kdw << 1));
      }
    }
    #pragma unroll
    for (int fm = 0; fm < 2; fm++)
      #pragma unroll
      for (int fn = 0; fn < 4; fn++)
        acc[fm][fn] = __builtin_amdgcn_wmma_f32_16x16x32_bf16(
            false, fa[fm].v, false, fb[fn].v, (short)0, acc[fm][fn], false, false);

    if (t + 1 < nk) l_store((t + 1) & 1);    // fill the other buffer for next iter
  }

  const size_t cbase = (size_t)bz * sC;
  #pragma unroll
  for (int fm = 0; fm < 2; fm++) {
    #pragma unroll
    for (int fn = 0; fn < 4; fn++) {
      int row0 = tileM + waveM * 32 + fm * 16 + lhi * 8;
      int col  = tileN + waveN * 64 + fn * 16 + l16;
      if (col >= N) continue;
      float bv = (flags & FB_BIAS) ? bias[col] : 0.f;
      #pragma unroll
      for (int i = 0; i < 8; i++) {
        int row = row0 + i;
        if (row >= M) continue;
        float v = acc[fm][fn][i] + bv;
        size_t idx = cbase + (size_t)row * ldc + col;
        if (flags & FB_ACC)  v += ((const float*)Cp)[idx];
        if (flags & FB_RELU) v = fmaxf(v, 0.f);
        if (flags & FB_OUTBF) ((unsigned short*)Cp)[idx] = f2bf(v);
        else                  ((float*)Cp)[idx] = v;
      }
    }
  }
}

// ---------------- softmax / window kernels ----------------

// scale [N(=b*L+s), L(=j)] f32 -> sc_win [b, j, 21]:
// full softmax over s per (b,j) column, then windowed renorm (win filled with 1e-10 off-window)
__global__ __launch_bounds__(256) void k_winsoftmax(const float* __restrict__ scale,
                                                    float* __restrict__ sc_win){
  int w = blockIdx.x * 8 + (threadIdx.x >> 5);   // over B*L anchors
  int lane = threadIdx.x & 31;
  int b = w >> 8, j = w & 255;
  const float* col = scale + ((size_t)(b << 8) << 8) + j;  // row (b*L+s), ldc=256
  float v[8], mx = -1e30f;
  #pragma unroll
  for (int i = 0; i < 8; i++) { int s = lane + (i << 5); v[i] = col[(size_t)s << 8]; mx = fmaxf(mx, v[i]); }
  for (int o = 16; o; o >>= 1) mx = fmaxf(mx, __shfl_xor(mx, o, 32));
  int lo = j - WIN_, hi = j + WIN_;
  float e[8], zs = 0.f, wsum = 0.f;
  #pragma unroll
  for (int i = 0; i < 8; i++) {
    int s = lane + (i << 5);
    e[i] = __expf(v[i] - mx);
    zs += e[i];
    if (s >= lo && s <= hi) wsum += e[i];
  }
  for (int o = 16; o; o >>= 1) { zs += __shfl_xor(zs, o, 32); wsum += __shfl_xor(wsum, o, 32); }
  float wn = wsum / zs;
  float denom = wn + 1e-10f * (1.0f - wn);
  float scl = 1.0f / (zs * denom);
  float* out = sc_win + (size_t)w * 21;
  if (lane < 21) out[lane] = 0.0f;
  #pragma unroll
  for (int i = 0; i < 8; i++) {
    int s = lane + (i << 5);
    if (s >= lo && s <= hi) out[s - j + WIN_] = e[i] * scl;
  }
}

// logits [B*L rows, 256] -> a_bf = softmax(tanh(.)) over last dim, bf16
__global__ __launch_bounds__(256) void k_attnsoftmax(const float* __restrict__ logits,
                                                     unsigned short* __restrict__ a_bf){
  int w = blockIdx.x * 8 + (threadIdx.x >> 5);
  int lane = threadIdx.x & 31;
  const float* row = logits + ((size_t)w << 8);
  float v[8], mx = -1e30f;
  #pragma unroll
  for (int i = 0; i < 8; i++) { v[i] = tanhf(row[lane + (i << 5)]); mx = fmaxf(mx, v[i]); }
  for (int o = 16; o; o >>= 1) mx = fmaxf(mx, __shfl_xor(mx, o, 32));
  float e[8], zs = 0.f;
  #pragma unroll
  for (int i = 0; i < 8; i++) { e[i] = __expf(v[i] - mx); zs += e[i]; }
  for (int o = 16; o; o >>= 1) zs += __shfl_xor(zs, o, 32);
  float inv = 1.0f / zs;
  unsigned short* out = a_bf + ((size_t)w << 8);
  #pragma unroll
  for (int i = 0; i < 8; i++) out[lane + (i << 5)] = f2bf(e[i] * inv);
}

// ---------------- graph gathers ----------------

// y_bf[k] = sum_{j in win(k), rel(j,k)==r} edge_norm(b,j,k) * x_bf[j]   (per conversation band)
__global__ __launch_bounds__(256) void k_rgcn_gather(const unsigned short* __restrict__ x_bf,
                                                     const float* __restrict__ sc_win,
                                                     const int* __restrict__ speakers,
                                                     unsigned short* __restrict__ y_bf, int r){
  int n = blockIdx.x;                  // dst node, n = b*L + k
  int b = n >> 8, k = n & 255;
  int tid = threadIdx.x;
  float a0 = 0.f, a1 = 0.f, a2 = 0.f, a3 = 0.f;
  int spk = speakers[k * B_ + b];
  int jlo = (k - WIN_ < 0) ? 0 : k - WIN_;
  int jhi = (k + WIN_ > 255) ? 255 : k + WIN_;
  for (int j = jlo; j <= jhi; j++) {
    int spj = speakers[j * B_ + b];
    int rel = spj * 4 + spk * 2 + ((j < k) ? 0 : 1);
    if (rel != r) continue;
    float en = sc_win[((size_t)((b << 8) + j)) * 21 + (k - j + WIN_)];
    const unsigned short* xr = x_bf + ((size_t)((b << 8) + j) << 10);
    a0 += en * bf2f(xr[tid]);
    a1 += en * bf2f(xr[tid + 256]);
    a2 += en * bf2f(xr[tid + 512]);
    a3 += en * bf2f(xr[tid + 768]);
  }
  unsigned short* yr = y_bf + ((size_t)n << 10);
  yr[tid] = f2bf(a0); yr[tid + 256] = f2bf(a1);
  yr[tid + 512] = f2bf(a2); yr[tid + 768] = f2bf(a3);
}

// nbr_bf[k] = sum_{j in win(k)} h1[j]  (unweighted)
__global__ __launch_bounds__(256) void k_nbr_gather(const float* __restrict__ h1,
                                                    unsigned short* __restrict__ nbr_bf){
  int n = blockIdx.x;
  int b = n >> 8, k = n & 255;
  int tid = threadIdx.x;
  float a0 = 0.f, a1 = 0.f;
  int jlo = (k - WIN_ < 0) ? 0 : k - WIN_;
  int jhi = (k + WIN_ > 255) ? 255 : k + WIN_;
  for (int j = jlo; j <= jhi; j++) {
    const float* hr = h1 + ((size_t)((b << 8) + j) << 9);
    a0 += hr[tid]; a1 += hr[tid + 256];
  }
  unsigned short* o = nbr_bf + ((size_t)n << 9);
  o[tid] = f2bf(a0); o[tid + 256] = f2bf(a1);
}

// ---------------- final FC + log_softmax (N=6, bandwidth-bound; plain VALU) ----------------
__global__ __launch_bounds__(256) void k_fc(const float* __restrict__ hidden,
                                            const float* __restrict__ Wfc,
                                            const float* __restrict__ bfc,
                                            float* __restrict__ out){
  int n = blockIdx.x * 256 + threadIdx.x;
  float a[C_];
  #pragma unroll
  for (int c = 0; c < C_; c++) a[c] = bfc[c];
  const float* h = hidden + ((size_t)n << 9);
  for (int k = 0; k < H_; k++) {
    float hv = h[k];
    const float* w = Wfc + k * C_;
    #pragma unroll
    for (int c = 0; c < C_; c++) a[c] += hv * w[c];
  }
  float m = a[0];
  #pragma unroll
  for (int c = 1; c < C_; c++) m = fmaxf(m, a[c]);
  float zs = 0.f;
  #pragma unroll
  for (int c = 0; c < C_; c++) zs += __expf(a[c] - m);
  float lse = __logf(zs);
  float* o = out + (size_t)n * C_;
  #pragma unroll
  for (int c = 0; c < C_; c++) o[c] = a[c] - m - lse;
}

// ---------------- host side ----------------

static void gemm(hipStream_t st, const void* A, const void* B, void* C, const float* bias,
                 int M, int N, int K, int lda, int ldb, int ldc,
                 long sA, long sB, long sC, int batch, int flags){
  dim3 g((N + BNT - 1) / BNT, (M + BMT - 1) / BMT, batch), blk(256);
  k_gemm<<<g, blk, 0, st>>>((const unsigned short*)A, (const unsigned short*)B, C, bias,
                            M, N, K, lda, ldb, ldc, sA, sB, sC, flags);
}

extern "C" void kernel_launch(void* const* d_in, const int* in_sizes, int n_in,
                              void* d_out, int out_size, void* d_ws, size_t ws_size,
                              hipStream_t stream) {
  (void)in_sizes; (void)n_in; (void)out_size;
  const float* features = (const float*)d_in[0];
  const float* Wscalar  = (const float*)d_in[1];
  const float* W_rel    = (const float*)d_in[2];
  const float* W_root   = (const float*)d_in[3];
  const float* b_rgcn   = (const float*)d_in[4];
  const float* W_nbr    = (const float*)d_in[5];
  const float* W_self   = (const float*)d_in[6];
  const float* b_gc     = (const float*)d_in[7];
  const float* W_match  = (const float*)d_in[8];
  const float* b_match  = (const float*)d_in[9];
  const float* W_lin    = (const float*)d_in[10];
  const float* b_lin    = (const float*)d_in[11];
  const float* W_fc     = (const float*)d_in[12];
  const float* b_fc     = (const float*)d_in[13];
  const int*   speakers = (const int*)d_in[14];

  char* wsb = (char*)d_ws;
  size_t off = 0;
  auto alloc = [&](size_t bytes) -> char* {
    char* p = wsb + off; off += (bytes + 255) & ~(size_t)255; return p;
  };

  unsigned short* x_bf     = (unsigned short*)alloc((size_t)N_ * D_ * 2);     // r0
  float*          scale    = (float*)alloc((size_t)N_ * L_ * 4);              // r1 (later: h1_bf)
  float*          sc_win   = (float*)alloc((size_t)B_ * L_ * 21 * 4);         // r2
  unsigned short* ws_bf    = (unsigned short*)alloc((size_t)D_ * L_ * 2);
  unsigned short* wrel_bf  = (unsigned short*)alloc((size_t)R_ * D_ * H_ * 2);
  unsigned short* wroot_bf = (unsigned short*)alloc((size_t)D_ * H_ * 2);
  unsigned short* wnbr_bf  = (unsigned short*)alloc((size_t)H_ * H_ * 2);
  unsigned short* wself_bf = (unsigned short*)alloc((size_t)H_ * H_ * 2);
  unsigned short* wmatch_bf= (unsigned short*)alloc((size_t)DH_ * DH_ * 2);
  unsigned short* wlin_bf  = (unsigned short*)alloc((size_t)DH_ * H_ * 2);
  float*          h1       = (float*)alloc((size_t)N_ * H_ * 4);              // r4 (later: xt_bf/att_bf, spans r4+r5)
  unsigned short* y_bf     = (unsigned short*)alloc((size_t)N_ * D_ * 2);     // r5
  unsigned short* nbr_bf   = (unsigned short*)alloc((size_t)N_ * H_ * 2);     // r6 (later: a_bf)
  float*          h2       = (float*)alloc((size_t)N_ * H_ * 4);              // r7 (later: hidden)
  unsigned short* em_bf    = (unsigned short*)alloc((size_t)N_ * DH_ * 2);    // r8
  float*          logits   = (float*)alloc((size_t)B_ * L_ * L_ * 4);         // r9
  if (off > ws_size) return;  // insufficient scratch

  unsigned short* h1_bf  = (unsigned short*)scale;   // scale dead after winsoftmax
  unsigned short* xt_bf  = (unsigned short*)h1;      // h1+y_bf dead by then (50.3MB < 67MB)
  unsigned short* att_bf = (unsigned short*)h1;      // xt dead after logits GEMM
  unsigned short* a_bf   = nbr_bf;                   // nbr dead after W_nbr GEMM
  float*          hidden = h2;                       // h2 dead after empack
  float*          out    = (float*)d_out;

  // weight conversions f32 -> bf16
  auto cvt = [&](const float* s, unsigned short* d, int n){
    k_f2bf<<<(n + 255) / 256, 256, 0, stream>>>(s, d, n);
  };
  cvt(Wscalar, ws_bf,    D_ * L_);
  cvt(W_rel,   wrel_bf,  R_ * D_ * H_);
  cvt(W_root,  wroot_bf, D_ * H_);
  cvt(W_nbr,   wnbr_bf,  H_ * H_);
  cvt(W_self,  wself_bf, H_ * H_);
  cvt(W_match, wmatch_bf,DH_ * DH_);
  cvt(W_lin,   wlin_bf,  DH_ * H_);

  // x (conversation-major) in bf16
  k_xpose<<<(N_ * D_) / 256, 256, 0, stream>>>(features, x_bf);

  // edge attention: scale = x @ Wscalar, then windowed softmax
  gemm(stream, x_bf, ws_bf, scale, nullptr, N_, L_, D_, D_, L_, L_, 0, 0, 0, 1, 0);
  k_winsoftmax<<<(B_ * L_) / 8, 256, 0, stream>>>(scale, sc_win);

  // RGCN: h1 = x@W_root + b_rgcn + sum_r (S_r x) @ W_rel[r]
  gemm(stream, x_bf, wroot_bf, h1, b_rgcn, N_, H_, D_, D_, H_, H_, 0, 0, 0, 1, FB_BIAS);
  for (int r = 0; r < R_; r++) {
    k_rgcn_gather<<<N_, 256, 0, stream>>>(x_bf, sc_win, speakers, y_bf, r);
    gemm(stream, y_bf, wrel_bf + (size_t)r * D_ * H_, h1, nullptr,
         N_, H_, D_, D_, H_, H_, 0, 0, 0, 1, FB_ACC);
  }

  // GraphConv: h2 = (window-sum h1)@W_nbr + h1@W_self + b_gc
  k_nbr_gather<<<N_, 256, 0, stream>>>(h1, nbr_bf);
  k_f2bf<<<(N_ * H_) / 256, 256, 0, stream>>>(h1, h1_bf, N_ * H_);
  gemm(stream, nbr_bf, wnbr_bf, h2, b_gc, N_, H_, H_, H_, H_, H_, 0, 0, 0, 1, FB_BIAS);
  gemm(stream, h1_bf, wself_bf, h2, nullptr, N_, H_, H_, H_, H_, H_, 0, 0, 0, 1, FB_ACC);

  // em = concat(x, h2), bf16
  k_empack<<<(N_ * DH_) / 256, 256, 0, stream>>>(x_bf, h2, em_bf);

  // MatchingAttention: xt = em@W_match + b_match (bf16 out)
  gemm(stream, em_bf, wmatch_bf, xt_bf, b_match, N_, DH_, DH_, DH_, DH_, DH_,
       0, 0, 0, 1, FB_BIAS | FB_OUTBF);
  // per-conversation logits = xt_b @ em_b^T
  gemm(stream, xt_bf, em_bf, logits, nullptr, L_, L_, DH_, DH_, DH_, L_,
       (long)L_ * DH_, (long)L_ * DH_, (long)L_ * L_, B_, FB_TRANSB);
  k_attnsoftmax<<<(B_ * L_) / 8, 256, 0, stream>>>(logits, a_bf);
  // att = a_b @ em_b (bf16 out)
  gemm(stream, a_bf, em_bf, att_bf, nullptr, L_, DH_, L_, L_, DH_, DH_,
       (long)L_ * L_, (long)L_ * DH_, (long)L_ * DH_, B_, FB_OUTBF);

  // hidden = relu(att@W_lin + b_lin)
  gemm(stream, att_bf, wlin_bf, hidden, b_lin, N_, H_, DH_, DH_, H_, H_,
       0, 0, 0, 1, FB_BIAS | FB_RELU);

  // final FC + log_softmax (output already conversation-major [N, C])
  k_fc<<<N_ / 256, 256, 0, stream>>>(hidden, W_fc, b_fc, out);
}
